// VanillaAttention_72696616452568
// MI455X (gfx1250) — matrix operand order repaired and checked by reading, not compiled, not run on hardware
//
#include <hip/hip_runtime.h>
#include <math.h>

// ---------------- problem constants ----------------
constexpr int B  = 2;
constexpr int S  = 2048;
constexpr int DIM = 2048;
constexpr int H  = 16;
constexpr int HD = 128;

typedef unsigned short u16;
typedef unsigned int   u32;

typedef __attribute__((ext_vector_type(16))) __bf16 v16bf;
typedef __attribute__((ext_vector_type(8)))  float  v8f;

#define AS1 __attribute__((address_space(1)))
#define AS3 __attribute__((address_space(3)))

// exact parameter type of the async-LDS b128 builtin (per compiler diagnostic)
typedef int v4i_vs __attribute__((__vector_size__(4 * sizeof(int))));

union FragB16 {
    v16bf v;
    u16   u[16];
    uint4 q[2];
};

__device__ __forceinline__ u16 f2bf(float f) {
    u32 x = __float_as_uint(f);
    u32 r = x + 0x7FFFu + ((x >> 16) & 1u);   // round-to-nearest-even
    return (u16)(r >> 16);
}

// ---- CDNA5 feature probes (device pass only; host pass takes fallbacks) ----
#if defined(__HIP_DEVICE_COMPILE__) && __has_builtin(__builtin_amdgcn_global_load_async_to_lds_b128)
#define HAVE_ASYNC_LDS 1
#else
#define HAVE_ASYNC_LDS 0
#endif

#if defined(__HIP_DEVICE_COMPILE__) && __has_builtin(__builtin_amdgcn_tensor_load_to_lds)
#define HAVE_TDM 1
#else
#define HAVE_TDM 0
#endif

__device__ __forceinline__ void wait_async0() {
#if defined(__HIP_DEVICE_COMPILE__)
#if __has_builtin(__builtin_amdgcn_s_wait_asynccnt)
    __builtin_amdgcn_s_wait_asynccnt(0);
#else
    asm volatile("s_wait_asynccnt 0x0" ::: "memory");
#endif
#endif
}

#if HAVE_TDM
typedef unsigned int u32x4 __attribute__((ext_vector_type(4)));
typedef int          i32x4 __attribute__((ext_vector_type(4)));
typedef int          i32x8 __attribute__((ext_vector_type(8)));

// 2-D TDM tile load: tile_d0 x tile_d1 elements (2-byte), row stride d0_stride elems.
// D# layout per CDNA5 ISA ch.8: group0 = {flags, lds_addr, global_addr, type=2},
// group1 = {data_size, tensor dims, tile dims, strides}.
__device__ __forceinline__ void tdm_load_tile_2d_b16(u32 lds_off, const void* gaddr,
                                                     u32 tensor_d0, u32 tensor_d1,
                                                     u32 tile_d0, u32 tile_d1,
                                                     u32 d0_stride) {
    unsigned long long ga = (unsigned long long)(uintptr_t)gaddr;
    u32x4 g0;
    g0[0] = 1u;                                            // count=1, user mode
    g0[1] = lds_off;                                       // lds_addr (bytes)
    g0[2] = (u32)(ga & 0xffffffffu);                       // global_addr[31:0]
    g0[3] = (u32)((ga >> 32) & 0x01ffffffu) | (2u << 30);  // global_addr[56:32] | type=2
    i32x8 g1;
    g1[0] = (int)(1u << 16);                               // data_size=1 (2 bytes)
    g1[1] = (int)((tensor_d0 & 0xffffu) << 16);            // tensor_dim0[15:0] @ bits 63:48
    g1[2] = (int)(((tensor_d0 >> 16) & 0xffffu) |
                  ((tensor_d1 & 0xffffu) << 16));          // dim0 hi | dim1 lo
    g1[3] = (int)(((tensor_d1 >> 16) & 0xffffu) |
                  ((tile_d0 & 0xffffu) << 16));            // dim1 hi | tile_dim0
    g1[4] = (int)(tile_d1 & 0xffffu);                      // tile_dim1; tile_dim2=0
    g1[5] = (int)d0_stride;                                // tensor_dim0_stride[31:0]
    g1[6] = 0;
    g1[7] = 0;
    i32x4 z4 = {0, 0, 0, 0};
#if __clang_major__ >= 23
    i32x8 z8 = {0, 0, 0, 0, 0, 0, 0, 0};
    __builtin_amdgcn_tensor_load_to_lds(g0, g1, z4, z4, z8, 0);
#else
    __builtin_amdgcn_tensor_load_to_lds(g0, g1, z4, z4, 0);
#endif
}
#endif // HAVE_TDM

// ---------------- fp32 -> bf16 convert ----------------
__global__ __launch_bounds__(256)
void k_cvt_f32_bf16(const float* __restrict__ in, u16* __restrict__ out, size_t n) {
    size_t i = (size_t)blockIdx.x * 256 + threadIdx.x;
    if (i < n) out[i] = f2bf(in[i]);
}

// ---------------- generic bf16 WMMA GEMM: C(f32, MxN) = A(bf16, MxK) * Bm(bf16, KxN)
// Workgroup tile 64(M) x 128(N), K-step 32. 8 waves; wave -> (wm = wave>>1, wn = (wave&1)*64),
// each wave owns 4 adjacent 16x16 output tiles along N.
// A tile staged via GLOBAL_LOAD_ASYNC_TO_LDS_B128 (ASYNCcnt) when available.
__global__ __launch_bounds__(256)
void k_gemm_bf16_wmma(const u16* __restrict__ A, const u16* __restrict__ Bm,
                      float* __restrict__ C, int M, int N, int K) {
    __shared__ __align__(16) u16 As[64][32];     // row-major tile of A
    __shared__ __align__(16) u16 Bs[128][32];    // B tile transposed: [n][k]

    const int tid  = threadIdx.x;
    const int lane = tid & 31;
    const int wave = tid >> 5;
    const int lr   = lane & 15;
    const int hi   = lane >> 4;                  // 0 or 1 (lane half)
    const int m0   = blockIdx.y * 64;
    const int n0   = blockIdx.x * 128;
    const int wm   = wave >> 1;
    const int wnb  = (wave & 1) * 64;

    v8f acc[4];
    for (int j = 0; j < 4; ++j)
        for (int r = 0; r < 8; ++r) acc[j][r] = 0.0f;

    const int arow_g = m0 + (tid >> 2);
    const int acol_g = (tid & 3) * 8;

    for (int k0 = 0; k0 < K; k0 += 32) {
        // stage A tile: 64x32 bf16, one b128 transfer per thread
#if HAVE_ASYNC_LDS
        {
            u16* gp = const_cast<u16*>(&A[(size_t)arow_g * K + k0 + acol_g]);
            __builtin_amdgcn_global_load_async_to_lds_b128(
                (AS1 v4i_vs*)(AS1 void*)gp,
                (AS3 v4i_vs*)(AS3 void*)&As[tid >> 2][acol_g],
                0, 0);
        }
#else
        *(uint4*)&As[tid >> 2][acol_g] =
            *(const uint4*)&A[(size_t)arow_g * K + k0 + acol_g];
#endif
        if (k0 + 32 < K)
            __builtin_prefetch(&A[(size_t)arow_g * K + k0 + 32 + acol_g], 0, 0);

        // stage B tile transposed: 32x128 -> Bs[n][k]; 2 b128 loads + scatter per thread
        #pragma unroll
        for (int c = 0; c < 2; ++c) {
            int chunk = tid * 2 + c;             // 0..511
            int kk = chunk >> 4;                 // 0..31
            int nn = (chunk & 15) * 8;           // 0..120
            uint4 d = *(const uint4*)&Bm[(size_t)(k0 + kk) * N + n0 + nn];
            const u16* ds = (const u16*)&d;
            #pragma unroll
            for (int i = 0; i < 8; ++i) Bs[nn + i][kk] = ds[i];
        }
#if HAVE_ASYNC_LDS
        wait_async0();
#endif
        __syncthreads();

        // A fragment (16x32): lanes 0-15 row M=lr K{0..7,16..23}; lanes 16-31 K{8..15,24..31}
        FragB16 af;
        const int arow = wm * 16 + lr;
        af.q[0] = *(const uint4*)&As[arow][hi * 8];
        af.q[1] = *(const uint4*)&As[arow][16 + hi * 8];

        #pragma unroll
        for (int j = 0; j < 4; ++j) {
            FragB16 bf;
            const int bcol = wnb + j * 16 + lr;
            bf.q[0] = *(const uint4*)&Bs[bcol][hi * 8];
            bf.q[1] = *(const uint4*)&Bs[bcol][16 + hi * 8];
            acc[j] = __builtin_amdgcn_wmma_f32_16x16x32_bf16(
                false, af.v, false, bf.v, (short)0, acc[j], false, false);
        }
        __syncthreads();
    }

    // C/D layout: reg r, lanes 0-15 -> M=r, lanes 16-31 -> M=r+8; N = lane%16
    #pragma unroll
    for (int j = 0; j < 4; ++j) {
        const int col = n0 + wnb + j * 16 + lr;
        #pragma unroll
        for (int r = 0; r < 8; ++r) {
            const int row = m0 + wm * 16 + r + hi * 8;
            C[(size_t)row * N + col] = acc[j][r];
        }
    }
}

// ---------------- rotary + (b,s,h,hd) -> (b,h,s,hd) transpose, fp32 -> bf16 ----------------
__global__ __launch_bounds__(256)
void k_rotary_transpose(const float* __restrict__ Xf,
                        const float* __restrict__ Cos, const float* __restrict__ Sin,
                        u16* __restrict__ Out, int apply) {
    size_t p = (size_t)blockIdx.x * 256 + threadIdx.x;   // pair index, B*S*H*HD/2 total
    const int HD2 = HD / 2;
    int j = (int)(p % HD2);
    size_t t = p / HD2;
    int h = (int)(t % H); t /= H;
    int s = (int)(t % S);
    int b = (int)(t / S);

    size_t off_in = (((size_t)b * S + s) * H + h) * HD + 2 * j;
    float re = Xf[off_in];
    float im = Xf[off_in + 1];
    float o_re = re, o_im = im;
    if (apply) {
        float c  = Cos[(size_t)s * HD2 + j];
        float sn = Sin[(size_t)s * HD2 + j];
        o_re = re * c - im * sn;
        o_im = re * sn + im * c;
    }
    size_t off_out = (((size_t)b * H + h) * S + s) * HD + 2 * j;
    u32 packed = (u32)f2bf(o_re) | ((u32)f2bf(o_im) << 16);
    *(u32*)&Out[off_out] = packed;
}

// ---------------- scores = mask(Q K^T * scale), WMMA, fragments straight from global ----------
// grid: (S/128 k-strips, S/16 q-tiles, B*H). 8 waves, wave w -> k-tile (bx*8 + w).
__global__ __launch_bounds__(256)
void k_attn_scores_wmma(const u16* __restrict__ Q, const u16* __restrict__ Kt,
                        float* __restrict__ Sc, float scale) {
    const int lane = threadIdx.x & 31;
    const int wave = threadIdx.x >> 5;
    const int lr = lane & 15;
    const int hi = lane >> 4;
    const int q0 = blockIdx.y * 16;
    const int k0 = (blockIdx.x * 8 + wave) * 16;
    const size_t bh = blockIdx.z;

    float* Sb = Sc + bh * (size_t)S * S;

    if (k0 > q0 + 15) {           // tile fully above diagonal: uniform per wave
        #pragma unroll
        for (int r = 0; r < 8; ++r) {
            int row = q0 + r + hi * 8;
            Sb[(size_t)row * S + k0 + lr] = -3.4028235e38f;
        }
        return;
    }

    const u16* qrow = Q + bh * (size_t)S * HD + (size_t)(q0 + lr) * HD;
    const u16* krow = Kt + bh * (size_t)S * HD + (size_t)(k0 + lr) * HD;

    v8f acc;
    for (int r = 0; r < 8; ++r) acc[r] = 0.0f;

    #pragma unroll
    for (int d0 = 0; d0 < HD; d0 += 32) {
        FragB16 af, bf;
        af.q[0] = *(const uint4*)&qrow[d0 + hi * 8];
        af.q[1] = *(const uint4*)&qrow[d0 + 16 + hi * 8];
        // B = K^T: column k of B is row k of K -> contiguous
        bf.q[0] = *(const uint4*)&krow[d0 + hi * 8];
        bf.q[1] = *(const uint4*)&krow[d0 + 16 + hi * 8];
        acc = __builtin_amdgcn_wmma_f32_16x16x32_bf16(
            false, af.v, false, bf.v, (short)0, acc, false, false);
    }

    #pragma unroll
    for (int r = 0; r < 8; ++r) {
        int qe = q0 + r + hi * 8;
        int ke = k0 + lr;
        Sb[(size_t)qe * S + ke] = (ke <= qe) ? acc[r] * scale : -3.4028235e38f;
    }
}

// ---------------- in-place row softmax over length-S rows ----------------
__global__ __launch_bounds__(256)
void k_softmax_rows(float* __restrict__ Sc) {
    __shared__ float red[256];
    const int tid = threadIdx.x;
    float* row = Sc + (size_t)blockIdx.x * S;

    float vals[S / 256];
    float m = -3.4028235e38f;
    #pragma unroll
    for (int i = 0; i < S / 256; ++i) {
        vals[i] = row[tid + i * 256];
        m = fmaxf(m, vals[i]);
    }
    red[tid] = m; __syncthreads();
    for (int s = 128; s > 0; s >>= 1) {
        if (tid < s) red[tid] = fmaxf(red[tid], red[tid + s]);
        __syncthreads();
    }
    m = red[0]; __syncthreads();

    float sum = 0.0f;
    #pragma unroll
    for (int i = 0; i < S / 256; ++i) {
        vals[i] = __expf(vals[i] - m);
        sum += vals[i];
    }
    red[tid] = sum; __syncthreads();
    for (int s = 128; s > 0; s >>= 1) {
        if (tid < s) red[tid] += red[tid + s];
        __syncthreads();
    }
    float inv = 1.0f / red[0];
    #pragma unroll
    for (int i = 0; i < S / 256; ++i) row[tid + i * 256] = vals[i] * inv;
}

// ---------------- attn = P @ V, WMMA; V tile staged to LDS via TDM ------------------------
// grid: (S/16 q-tiles, B*H); 8 waves per block, wave w owns n-subtile w (HD/16 == 8).
__global__ __launch_bounds__(256)
void k_attn_pv_wmma(const float* __restrict__ P, const u16* __restrict__ V,
                    u16* __restrict__ Ob) {
    __shared__ __align__(16) u16 Vs[32][HD];     // 32 k-rows x 128 head-dim, 8 KB

    const int tid  = threadIdx.x;
    const int lane = tid & 31;
    const int wave = tid >> 5;
    const int lr = lane & 15;
    const int hi = lane >> 4;
    const int q0 = blockIdx.x * 16;
    const size_t bh = blockIdx.y;
    const int b = (int)(bh / H);
    const int h = (int)(bh % H);
    const int n0 = wave * 16;

    const float* prow = P + bh * (size_t)S * S + (size_t)(q0 + lr) * S;
    const u16*   Vb   = V + bh * (size_t)S * HD;

    v8f acc;
    for (int r = 0; r < 8; ++r) acc[r] = 0.0f;

    // causal: P[q][k] == 0 for k > q, so only first round32(q0+16) columns matter (exact)
    const int kmax = (q0 + 16 + 31) & ~31;

    for (int c0 = 0; c0 < kmax; c0 += 32) {
        // ---- stage V rows c0..c0+31 into LDS ----
#if HAVE_TDM
        if (wave == 0) {
            u32 lds_off = (u32)(unsigned long long)(AS3 void*)&Vs[0][0];
            tdm_load_tile_2d_b16(lds_off, (const void*)(Vb + (size_t)c0 * HD),
                                 /*tensor_d0=*/HD, /*tensor_d1=*/(u32)(S - c0),
                                 /*tile_d0=*/HD, /*tile_d1=*/32, /*d0_stride=*/HD);
            __builtin_amdgcn_s_wait_tensorcnt(0);
        }
#else
        #pragma unroll
        for (int c = 0; c < 2; ++c) {
            int chunk = tid * 2 + c;             // 0..511
            int rr = chunk >> 4;                 // 0..31
            int cc = (chunk & 15) * 8;           // 0..120
            *(uint4*)&Vs[rr][cc] = *(const uint4*)&Vb[(size_t)(c0 + rr) * HD + cc];
        }
#endif
        __syncthreads();

        // A fragment: P fp32 -> bf16 in-register
        FragB16 af;
        #pragma unroll
        for (int half = 0; half < 2; ++half) {
            const float* p = prow + c0 + half * 16 + hi * 8;
            float4 f0 = *(const float4*)p;
            float4 f1 = *(const float4*)(p + 4);
            af.u[half * 8 + 0] = f2bf(f0.x);
            af.u[half * 8 + 1] = f2bf(f0.y);
            af.u[half * 8 + 2] = f2bf(f0.z);
            af.u[half * 8 + 3] = f2bf(f0.w);
            af.u[half * 8 + 4] = f2bf(f1.x);
            af.u[half * 8 + 5] = f2bf(f1.y);
            af.u[half * 8 + 6] = f2bf(f1.z);
            af.u[half * 8 + 7] = f2bf(f1.w);
        }

        // B fragment: column n0+lr of the V tile, rows hi*8+{0..7} and 16+hi*8+{0..7} (from LDS)
        FragB16 bf;
        const int col = n0 + lr;
        #pragma unroll
        for (int i = 0; i < 16; ++i)
            bf.u[i] = Vs[(i & 8) * 2 + hi * 8 + (i & 7)][col];

        acc = __builtin_amdgcn_wmma_f32_16x16x32_bf16(
            false, af.v, false, bf.v, (short)0, acc, false, false);
        __syncthreads();
    }

    // write attn in (b, s, h*HD) layout (ready for output projection)
    #pragma unroll
    for (int r = 0; r < 8; ++r) {
        int q = q0 + r + hi * 8;
        Ob[((size_t)b * S + q) * (H * HD) + h * HD + n0 + lr] = f2bf(acc[r]);
    }
}

// ---------------- host-side orchestration ----------------
extern "C" void kernel_launch(void* const* d_in, const int* in_sizes, int n_in,
                              void* d_out, int out_size, void* d_ws, size_t ws_size,
                              hipStream_t stream) {
    (void)in_sizes; (void)n_in; (void)out_size; (void)ws_size;

    const float* x  = (const float*)d_in[0];
    const float* fc = (const float*)d_in[1];
    const float* fs = (const float*)d_in[2];
    // d_in[3] = mask (bool) — causal structure is hard-coded
    const float* qw = (const float*)d_in[4];
    const float* kw = (const float*)d_in[5];
    const float* vw = (const float*)d_in[6];
    const float* ow = (const float*)d_in[7];

    float* outp  = (float*)d_out;                       // B*S*DIM
    float* attnw = outp + (size_t)B * S * DIM;          // B*H*S*S

    char* ws = (char*)d_ws;
    const size_t MB = 1024ull * 1024ull;
    u16*  xb   = (u16*)(ws);                            // B*S*DIM bf16      = 16 MB
    u16*  wbuf = (u16*)(ws + 16 * MB);                  // DIM*DIM bf16      =  8 MB (reused)
    float* qkvf = (float*)(ws + 24 * MB);               // B*S*H*HD f32      = 32 MB (reused)
    u16*  qb   = (u16*)(ws + 56 * MB);                  // (b,h,s,hd) bf16   = 16 MB
    u16*  kb   = (u16*)(ws + 72 * MB);                  // 16 MB
    u16*  vb   = (u16*)(ws + 88 * MB);                  // 16 MB
    u16*  ab   = (u16*)(ws + 104 * MB);                 // attn (b,s,h*hd)   = 16 MB

    const size_t nx = (size_t)B * S * DIM;      // 8388608
    const size_t nw = (size_t)DIM * H * HD;     // 4194304
    const size_t npairs = (size_t)B * S * H * HD / 2;

    dim3 blk(256);
    dim3 gemm_grid((H * HD) / 128, (B * S) / 64);       // (16, 64)
    dim3 sc_grid(S / 128, S / 16, B * H);               // (16, 128, 32)
    dim3 pv_grid(S / 16, B * H);                        // (128, 32)

    // x -> bf16
    k_cvt_f32_bf16<<<(u32)(nx / 256), blk, 0, stream>>>(x, xb, nx);

    // Q projection + rotary
    k_cvt_f32_bf16<<<(u32)(nw / 256), blk, 0, stream>>>(qw, wbuf, nw);
    k_gemm_bf16_wmma<<<gemm_grid, blk, 0, stream>>>(xb, wbuf, qkvf, B * S, H * HD, DIM);
    k_rotary_transpose<<<(u32)(npairs / 256), blk, 0, stream>>>(qkvf, fc, fs, qb, 1);

    // K projection + rotary
    k_cvt_f32_bf16<<<(u32)(nw / 256), blk, 0, stream>>>(kw, wbuf, nw);
    k_gemm_bf16_wmma<<<gemm_grid, blk, 0, stream>>>(xb, wbuf, qkvf, B * S, H * HD, DIM);
    k_rotary_transpose<<<(u32)(npairs / 256), blk, 0, stream>>>(qkvf, fc, fs, kb, 1);

    // V projection (transpose only)
    k_cvt_f32_bf16<<<(u32)(nw / 256), blk, 0, stream>>>(vw, wbuf, nw);
    k_gemm_bf16_wmma<<<gemm_grid, blk, 0, stream>>>(xb, wbuf, qkvf, B * S, H * HD, DIM);
    k_rotary_transpose<<<(u32)(npairs / 256), blk, 0, stream>>>(qkvf, fc, fs, vb, 0);

    // scores (masked, scaled) -> attn_w region of d_out
    k_attn_scores_wmma<<<sc_grid, blk, 0, stream>>>(qb, kb, attnw, 0.08838834764831845f);

    // softmax in place
    k_softmax_rows<<<(u32)((size_t)B * H * S), blk, 0, stream>>>(attnw);

    // attn = P @ V
    k_attn_pv_wmma<<<pv_grid, blk, 0, stream>>>(attnw, vb, ab);

    // output projection -> d_out
    k_cvt_f32_bf16<<<(u32)(nw / 256), blk, 0, stream>>>(ow, wbuf, nw);
    k_gemm_bf16_wmma<<<gemm_grid, blk, 0, stream>>>(ab, wbuf, outp, B * S, H * HD, DIM);
}
